// HierarchicalLSTM_70531952935144
// MI455X (gfx1250) — matrix operand order repaired and checked by reading, not compiled, not run on hardware
//
#include <hip/hip_runtime.h>

// ---------------------------------------------------------------------------
// MI455X (gfx1250) hierarchical LSTM prefetcher.
//  - Attention algebraically fused (never materializes k/v): 137 -> ~13 GFLOP.
//  - LSTM input projections hoisted into one big WMMA GEMM per layer; only
//    h@U (0.27 GFLOP/step) is sequential.
//  - All GEMMs: bf16 inputs, f32 accumulate via v_wmma_f32_16x16x32_bf16.
//  - Register blocking: generic GEMM uses a 32x64 tile per wave (8 WMMAs per
//    2 A-frag + 4 B-frag loads); lstm_step feeds 8 WMMAs per A-frag.
// ---------------------------------------------------------------------------

typedef __bf16 bf16;
typedef __attribute__((ext_vector_type(16))) __bf16 v16bf;
typedef __attribute__((ext_vector_type(8)))  __bf16 v8bf;
typedef __attribute__((ext_vector_type(8)))  float  v8f;

constexpr int kS    = 64;      // sequence length
constexpr int kB    = 512;     // batch
constexpr int kH    = 256;     // hidden
constexpr int kEPC  = 64;      // pc embed dim
constexpr int kEPG  = 256;     // page embed dim (== DK)
constexpr int kNEXP = 16;      // experts
constexpr int kDIN  = 576;     // 64 + 256 + 256
constexpr int kPGV  = 100000;  // page vocab
constexpr int kOFFV = 64;      // offset vocab
constexpr int kTOK  = kB * kS; // 32768 tokens

__device__ __forceinline__ v8f wmma_bf16f32(v16bf a, v16bf b, v8f c) {
  return __builtin_amdgcn_wmma_f32_16x16x32_bf16(false, a, false, b,
                                                 (short)0, c, false, false);
}

// A-fragment (16x32 bf16) from a row-major tile with row stride `stride`.
// Lane L: m = L&15, half = L>>4. element j -> K = k0 + 16*half + j
// => two contiguous 16-byte chunks at +16*half and +16*half+8.
__device__ __forceinline__ v16bf load_a_frag(const bf16* base, int stride, int k0) {
  const int lane = threadIdx.x & 31;
  const int m = lane & 15, hh = lane >> 4;
  const v8bf* p = (const v8bf*)(base + (size_t)m * stride + k0 + hh * 16);
  v16bf a;
  ((v8bf*)&a)[0] = p[0];   // K = k0+16h .. +7
  ((v8bf*)&a)[1] = p[1];   // K = k0+16h+8 .. +7
  return a;
}

__device__ __forceinline__ float sigm(float x) {
  return 1.f / (1.f + __expf(-x));
}
__device__ __forceinline__ float tanh_f(float x) {
  x = fminf(fmaxf(x, -15.f), 15.f);
  const float e = __expf(2.f * x);
  return (e - 1.f) / (e + 1.f);
}

// ---------------------------------------------------------------------------
// Weight prep kernels (run once per call; tiny fraction of total work)
// ---------------------------------------------------------------------------
__global__ void cvt_bf16(const float* __restrict__ src, bf16* __restrict__ dst, int n) {
  const int i = blockIdx.x * 256 + threadIdx.x;
  if (i < n) dst[i] = (bf16)src[i];
}

// src (R x C) row-major f32 -> dst (C x R) row-major bf16 (transposed)
__global__ void cvt_bf16_T(const float* __restrict__ src, bf16* __restrict__ dst,
                           int R, int C) {
  const int i = blockIdx.x * 256 + threadIdx.x;
  if (i < R * C) {
    const int r = i / C, c = i - r * C;
    dst[(size_t)c * R + r] = (bf16)src[i];
  }
}

// bvo[n] = (bv @ Wo)[n] + bo[n]
__global__ void compute_bvo(const float* __restrict__ bv, const float* __restrict__ Wo,
                            const float* __restrict__ bo, float* __restrict__ bvo) {
  const int n = threadIdx.x;  // 256 threads
  float s = bo[n];
  for (int j = 0; j < kEPG; ++j) s += bv[j] * Wo[j * kEPG + n];
  bvo[n] = s;
}

// ---------------------------------------------------------------------------
// Generic WMMA GEMM:  C[M,N] = A[M,K](bf16, row stride ldA) @ B + bias
// B supplied transposed: BT[N,K] bf16 (row stride ldBT).
// Block = 256 threads (8 waves). Block tile: 32 (M) x 512 (N).
// Wave w owns a 32x64 tile (2 M-subtiles x 4 N-subtiles -> 8 accumulators):
// per K-step, 2 A-frags (LDS) + 4 B-frags (global) feed 8 WMMAs.
// grid.y = M/32, grid.x = ceil(N/512). K % 32 == 0, N % 16 == 0.
// ---------------------------------------------------------------------------
__global__ __launch_bounds__(256) void gemm_bf16(
    const bf16* __restrict__ A, int ldA,
    const bf16* __restrict__ BT, int ldBT,
    float* __restrict__ C, int ldC,
    const float* __restrict__ bias,
    int N, int K) {
  __shared__ bf16 As[32 * 576];  // sized for max K (576)
  const int m0 = blockIdx.y * 32;
  const int tid = threadIdx.x;
  for (int idx = tid; idx < 32 * K; idx += 256) {
    const int r = idx / K, c = idx - r * K;
    As[idx] = A[(size_t)(m0 + r) * ldA + c];
  }
  __syncthreads();

  const int w = tid >> 5, lane = tid & 31;
  const int n0 = blockIdx.x * 512 + w * 64;
  if (n0 >= N) return;
  const int n = lane & 15, kh = lane >> 4;

  // Clamp invalid N-subtiles onto subtile 0 so loads stay in-bounds; the
  // store guard skips them. N % 16 == 0 always.
  const bf16* bt[4];
#pragma unroll
  for (int ni = 0; ni < 4; ++ni) {
    const int nr = (n0 + ni * 16 < N) ? (n0 + ni * 16) : n0;
    bt[ni] = BT + (size_t)(nr + n) * ldBT + kh * 16;
  }

  v8f acc[2][4];
#pragma unroll
  for (int mi = 0; mi < 2; ++mi)
#pragma unroll
    for (int ni = 0; ni < 4; ++ni)
      acc[mi][ni] = (v8f){0.f, 0.f, 0.f, 0.f, 0.f, 0.f, 0.f, 0.f};

  for (int k0 = 0; k0 < K; k0 += 32) {
    const v16bf a0 = load_a_frag(As, K, k0);
    const v16bf a1 = load_a_frag(As + 16 * K, K, k0);
    __builtin_prefetch(bt[0] + k0 + 128, 0, 1);  // stream next B chunk
    v16bf b[4];
#pragma unroll
    for (int ni = 0; ni < 4; ++ni) b[ni] = *(const v16bf*)(bt[ni] + k0);
#pragma unroll
    for (int ni = 0; ni < 4; ++ni) {
      acc[0][ni] = wmma_bf16f32(a0, b[ni], acc[0][ni]);
      acc[1][ni] = wmma_bf16f32(a1, b[ni], acc[1][ni]);
    }
  }

#pragma unroll
  for (int ni = 0; ni < 4; ++ni) {
    const int nc = n0 + ni * 16 + n;
    if (n0 + ni * 16 >= N) continue;
    const float bs = bias ? bias[nc] : 0.f;
#pragma unroll
    for (int mi = 0; mi < 2; ++mi)
#pragma unroll
      for (int r = 0; r < 8; ++r)
        C[(size_t)(m0 + mi * 16 + r + 8 * kh) * ldC + nc] = acc[mi][ni][r] + bs;
  }
}

// ---------------------------------------------------------------------------
// Feature builder: embeddings + fused attention for 16 tokens per block.
// xfeat row = [pc(64) | pg(256) | o(256)] bf16.
//   Q   = PG @ Wq + bq                    (WMMA)
//   U   = Q @ Wk^T                        (WMMA)
//   s_n = (U_tok . off_n) / 16 ; attn = softmax ; mix = sum attn_n off_n
//   O   = MIX @ (Wv Wo) + (bv Wo + bo)    (WMMA)
// Each wave owns N-tiles {w*16, 128+w*16}; one A-frag feeds both WMMAs.
// ---------------------------------------------------------------------------
__global__ __launch_bounds__(256) void build_features(
    const int* __restrict__ inputs, const float* __restrict__ pc_emb,
    const float* __restrict__ page_emb, const float* __restrict__ off_emb,
    const bf16* __restrict__ WqT, const float* __restrict__ bq,
    const bf16* __restrict__ WkB, const bf16* __restrict__ WvoT,
    const float* __restrict__ bvo, bf16* __restrict__ xfeat) {
  __shared__ bf16  PG[16 * 256];
  __shared__ bf16  Qb[16 * 256];
  __shared__ float Uf[16 * 256];
  __shared__ bf16  MIX[16 * 256];
  __shared__ int pci[16], pgi[16], offi[16];

  const int g = blockIdx.x;  // token group (2048 groups)
  const int tid = threadIdx.x, w = tid >> 5, lane = tid & 31;
  if (tid < 16) {
    const int tok = g * 16 + tid;
    const int b = tok >> 6, s = tok & 63;
    const int* row = inputs + b * (3 * kS);
    pci[tid]  = row[s];
    pgi[tid]  = row[kS + s];
    offi[tid] = row[2 * kS + s];
  }
  __syncthreads();

  for (int idx = tid; idx < 16 * 256; idx += 256) {
    const int i = idx >> 8, c = idx & 255;
    const float v = page_emb[(size_t)pgi[i] * kEPG + c];
    PG[idx] = (bf16)v;
    xfeat[(size_t)(g * 16 + i) * kDIN + kEPC + c] = (bf16)v;
  }
  for (int idx = tid; idx < 16 * 64; idx += 256) {
    const int i = idx >> 6, c = idx & 63;
    xfeat[(size_t)(g * 16 + i) * kDIN + c] = (bf16)pc_emb[(size_t)pci[i] * kEPC + c];
  }
  __syncthreads();

  const int n = lane & 15, kh = lane >> 4;
  const int n0a = w * 16, n0b = 128 + w * 16;

  // Q = PG @ Wq + bq  -> Qb (bf16)
  {
    v8f acc0 = {0.f,0.f,0.f,0.f,0.f,0.f,0.f,0.f}, acc1 = acc0;
    const bf16* bt0 = WqT + (size_t)(n0a + n) * 256 + kh * 16;
    const bf16* bt1 = WqT + (size_t)(n0b + n) * 256 + kh * 16;
    for (int k0 = 0; k0 < 256; k0 += 32) {
      const v16bf a = load_a_frag(PG, 256, k0);
      acc0 = wmma_bf16f32(a, *(const v16bf*)(bt0 + k0), acc0);
      acc1 = wmma_bf16f32(a, *(const v16bf*)(bt1 + k0), acc1);
    }
    const float bq0 = bq[n0a + n], bq1 = bq[n0b + n];
#pragma unroll
    for (int r = 0; r < 8; ++r) {
      Qb[(r + 8 * kh) * 256 + n0a + n] = (bf16)(acc0[r] + bq0);
      Qb[(r + 8 * kh) * 256 + n0b + n] = (bf16)(acc1[r] + bq1);
    }
  }
  __syncthreads();

  // U = Q @ Wk^T  (BT = Wk row-major) -> Uf (f32)
  {
    v8f acc0 = {0.f,0.f,0.f,0.f,0.f,0.f,0.f,0.f}, acc1 = acc0;
    const bf16* bt0 = WkB + (size_t)(n0a + n) * 256 + kh * 16;
    const bf16* bt1 = WkB + (size_t)(n0b + n) * 256 + kh * 16;
    for (int k0 = 0; k0 < 256; k0 += 32) {
      const v16bf a = load_a_frag(Qb, 256, k0);
      acc0 = wmma_bf16f32(a, *(const v16bf*)(bt0 + k0), acc0);
      acc1 = wmma_bf16f32(a, *(const v16bf*)(bt1 + k0), acc1);
    }
#pragma unroll
    for (int r = 0; r < 8; ++r) {
      Uf[(r + 8 * kh) * 256 + n0a + n] = acc0[r];
      Uf[(r + 8 * kh) * 256 + n0b + n] = acc1[r];
    }
  }
  __syncthreads();

  // Per-token scores / softmax / mix. Wave w handles tokens 2w, 2w+1.
  for (int i = w * 2; i < w * 2 + 2; ++i) {
    const float* offp = off_emb + (size_t)offi[i] * (kNEXP * kEPG);
    float u8[8];
#pragma unroll
    for (int k = 0; k < 8; ++k) u8[k] = Uf[i * 256 + lane * 8 + k];
    float sc[16];
#pragma unroll
    for (int e = 0; e < 16; ++e) {
      const float* oe = offp + e * 256 + lane * 8;
      float part = 0.f;
#pragma unroll
      for (int k = 0; k < 8; ++k) part += oe[k] * u8[k];
#pragma unroll
      for (int d = 16; d >= 1; d >>= 1) part += __shfl_xor(part, d, 32);
      sc[e] = part * 0.0625f;  // 1/sqrt(DK) = 1/16
    }
    float mx = sc[0];
#pragma unroll
    for (int e = 1; e < 16; ++e) mx = fmaxf(mx, sc[e]);
    float sum = 0.f;
#pragma unroll
    for (int e = 0; e < 16; ++e) { sc[e] = __expf(sc[e] - mx); sum += sc[e]; }
    const float inv = 1.f / sum;
    float mix8[8] = {0.f, 0.f, 0.f, 0.f, 0.f, 0.f, 0.f, 0.f};
#pragma unroll
    for (int e = 0; e < 16; ++e) {
      const float at = sc[e] * inv;
      const float* oe = offp + e * 256 + lane * 8;
#pragma unroll
      for (int k = 0; k < 8; ++k) mix8[k] += at * oe[k];
    }
#pragma unroll
    for (int k = 0; k < 8; ++k) MIX[i * 256 + lane * 8 + k] = (bf16)mix8[k];
  }
  __syncthreads();

  // O = MIX @ Wvo + bvo -> xfeat[:, 320:576]
  {
    v8f acc0 = {0.f,0.f,0.f,0.f,0.f,0.f,0.f,0.f}, acc1 = acc0;
    const bf16* bt0 = WvoT + (size_t)(n0a + n) * 256 + kh * 16;
    const bf16* bt1 = WvoT + (size_t)(n0b + n) * 256 + kh * 16;
    for (int k0 = 0; k0 < 256; k0 += 32) {
      const v16bf a = load_a_frag(MIX, 256, k0);
      acc0 = wmma_bf16f32(a, *(const v16bf*)(bt0 + k0), acc0);
      acc1 = wmma_bf16f32(a, *(const v16bf*)(bt1 + k0), acc1);
    }
    const float bv0 = bvo[n0a + n], bv1 = bvo[n0b + n];
#pragma unroll
    for (int r = 0; r < 8; ++r) {
      const size_t rowr = (size_t)(g * 16 + r + 8 * kh) * kDIN + (kEPC + kEPG);
      xfeat[rowr + n0a + n] = (bf16)(acc0[r] + bv0);
      xfeat[rowr + n0b + n] = (bf16)(acc1[r] + bv1);
    }
  }
}

// ---------------------------------------------------------------------------
// One LSTM recurrent step (layer-agnostic):
//   z = zbuf[:,t,:]  (precomputed x@W + b)  +  h_{t-1} @ U
//   gates -> update cstate (f32), write h_t to hs (bf16, [B][S][H]).
// grid.x = B/16; block = 8 waves. Wave w covers hidden cols {w*16, 128+w*16}
// for all 4 gates: one A-frag feeds 8 WMMAs per K-step (B pointers differ by
// immediate-foldable offsets).
// ---------------------------------------------------------------------------
__global__ __launch_bounds__(256) void lstm_step(
    const bf16* __restrict__ hs, const bf16* __restrict__ UT,
    const float* __restrict__ zbuf, float* __restrict__ cstate,
    bf16* __restrict__ hs_out, int t) {
  __shared__ bf16 Hs[16 * 256];
  const int m0 = blockIdx.x * 16;
  const int tid = threadIdx.x;
  if (t > 0) {
    for (int idx = tid; idx < 16 * 256; idx += 256) {
      const int r = idx >> 8, c = idx & 255;
      Hs[idx] = hs[((size_t)(m0 + r) * kS + (t - 1)) * kH + c];
    }
  }
  __syncthreads();

  const int w = tid >> 5, lane = tid & 31;
  const int n = lane & 15, kh = lane >> 4;

  v8f acc[8];  // [pass*4 + gate]
#pragma unroll
  for (int q = 0; q < 8; ++q)
    acc[q] = (v8f){0.f, 0.f, 0.f, 0.f, 0.f, 0.f, 0.f, 0.f};

  if (t > 0) {
    const bf16* base = UT + (size_t)(w * 16 + n) * 256 + kh * 16;
    for (int k0 = 0; k0 < 256; k0 += 32) {
      const v16bf a = load_a_frag(Hs, 256, k0);
#pragma unroll
      for (int p = 0; p < 2; ++p)
#pragma unroll
        for (int gg = 0; gg < 4; ++gg) {
          const bf16* bp = base + (size_t)(gg * 256 + p * 128) * 256 + k0;
          acc[p * 4 + gg] = wmma_bf16f32(a, *(const v16bf*)bp, acc[p * 4 + gg]);
        }
    }
  }

#pragma unroll
  for (int p = 0; p < 2; ++p) {
    const int j = p * 128 + w * 16 + n;
#pragma unroll
    for (int r = 0; r < 8; ++r) {
      const int mm = m0 + r + 8 * kh;
      const size_t zrow = ((size_t)mm * kS + t) * (4 * kH);
      const float zi = acc[p * 4 + 0][r] + zbuf[zrow + j];
      const float zf = acc[p * 4 + 1][r] + zbuf[zrow + kH + j];
      const float zg = acc[p * 4 + 2][r] + zbuf[zrow + 2 * kH + j];
      const float zo = acc[p * 4 + 3][r] + zbuf[zrow + 3 * kH + j];
      const float cp = (t > 0) ? cstate[(size_t)mm * kH + j] : 0.f;
      const float cn = sigm(zf) * cp + sigm(zi) * tanh_f(zg);
      const float hn = sigm(zo) * tanh_f(cn);
      cstate[(size_t)mm * kH + j] = cn;
      hs_out[((size_t)mm * kS + t) * kH + j] = (bf16)hn;
    }
  }
}

// ---------------------------------------------------------------------------
// Host-side orchestration (all on `stream`, graph-capture safe).
// ---------------------------------------------------------------------------
extern "C" void kernel_launch(void* const* d_in, const int* in_sizes, int n_in,
                              void* d_out, int out_size, void* d_ws, size_t ws_size,
                              hipStream_t stream) {
  (void)in_sizes; (void)n_in; (void)out_size; (void)ws_size;
  const int*   inputs   = (const int*)  d_in[0];
  const float* pc_emb   = (const float*)d_in[1];
  const float* page_emb = (const float*)d_in[2];
  const float* off_emb  = (const float*)d_in[3];
  const float* Wq  = (const float*)d_in[4];   const float* bq  = (const float*)d_in[5];
  const float* Wk  = (const float*)d_in[6];   /* bk (d_in[7]) cancels in softmax */
  const float* Wv  = (const float*)d_in[8];   const float* bv  = (const float*)d_in[9];
  const float* Wo  = (const float*)d_in[10];  const float* bo  = (const float*)d_in[11];
  const float* cW0 = (const float*)d_in[12];  const float* cU0 = (const float*)d_in[13];
  const float* cb0 = (const float*)d_in[14];
  const float* cW1 = (const float*)d_in[15];  const float* cU1 = (const float*)d_in[16];
  const float* cb1 = (const float*)d_in[17];
  const float* fW0 = (const float*)d_in[18];  const float* fU0 = (const float*)d_in[19];
  const float* fb0 = (const float*)d_in[20];
  const float* fW1 = (const float*)d_in[21];  const float* fU1 = (const float*)d_in[22];
  const float* fb1 = (const float*)d_in[23];
  const float* Wpage = (const float*)d_in[24]; const float* bpage = (const float*)d_in[25];
  const float* Woff  = (const float*)d_in[26]; const float* boff  = (const float*)d_in[27];
  float* out = (float*)d_out;

  char* ws = (char*)d_ws;
  size_t pos = 0;
  auto alloc = [&](size_t bytes) -> void* {
    pos = (pos + 255) & ~(size_t)255;
    void* p = ws + pos;
    pos += bytes;
    return p;
  };

  bf16*  WqT     = (bf16*) alloc((size_t)kEPG * kEPG * 2);
  bf16*  WkB     = (bf16*) alloc((size_t)kEPG * kEPG * 2);   // Wk row-major bf16
  bf16*  WvB     = (bf16*) alloc((size_t)kEPG * kEPG * 2);   // Wv row-major bf16
  bf16*  WoT     = (bf16*) alloc((size_t)kEPG * kEPG * 2);
  float* WvoT_f  = (float*)alloc((size_t)kEPG * kEPG * 4);
  bf16*  WvoT    = (bf16*) alloc((size_t)kEPG * kEPG * 2);
  float* bvo     = (float*)alloc((size_t)kEPG * 4);
  bf16*  cW0T    = (bf16*) alloc((size_t)kDIN * 4 * kH * 2);
  bf16*  cU0T    = (bf16*) alloc((size_t)kH * 4 * kH * 2);
  bf16*  cW1T    = (bf16*) alloc((size_t)kH * 4 * kH * 2);
  bf16*  cU1T    = (bf16*) alloc((size_t)kH * 4 * kH * 2);
  bf16*  fW0T    = (bf16*) alloc((size_t)kDIN * 4 * kH * 2);
  bf16*  fU0T    = (bf16*) alloc((size_t)kH * 4 * kH * 2);
  bf16*  fW1T    = (bf16*) alloc((size_t)kH * 4 * kH * 2);
  bf16*  fU1T    = (bf16*) alloc((size_t)kH * 4 * kH * 2);
  bf16*  WpageT  = (bf16*) alloc((size_t)kPGV * kH * 2);
  bf16*  WoffT   = (bf16*) alloc((size_t)kOFFV * kH * 2);
  bf16*  xfeat   = (bf16*) alloc((size_t)kTOK * kDIN * 2);
  float* zbuf    = (float*)alloc((size_t)kTOK * 4 * kH * 4);
  bf16*  hs0     = (bf16*) alloc((size_t)kTOK * kH * 2);
  bf16*  hs1     = (bf16*) alloc((size_t)kTOK * kH * 2);
  float* cst     = (float*)alloc((size_t)kB * kH * 4);

  auto cdiv = [](int a, int b) { return (a + b - 1) / b; };

  // ---- one-time weight prep -------------------------------------------------
  cvt_bf16_T<<<cdiv(kEPG * kEPG, 256), 256, 0, stream>>>(Wq, WqT, kEPG, kEPG);
  cvt_bf16  <<<cdiv(kEPG * kEPG, 256), 256, 0, stream>>>(Wk, WkB, kEPG * kEPG);
  cvt_bf16  <<<cdiv(kEPG * kEPG, 256), 256, 0, stream>>>(Wv, WvB, kEPG * kEPG);
  cvt_bf16_T<<<cdiv(kEPG * kEPG, 256), 256, 0, stream>>>(Wo, WoT, kEPG, kEPG);
  cvt_bf16_T<<<cdiv(kDIN * 4 * kH, 256), 256, 0, stream>>>(cW0, cW0T, kDIN, 4 * kH);
  cvt_bf16_T<<<cdiv(kH * 4 * kH, 256), 256, 0, stream>>>(cU0, cU0T, kH, 4 * kH);
  cvt_bf16_T<<<cdiv(kH * 4 * kH, 256), 256, 0, stream>>>(cW1, cW1T, kH, 4 * kH);
  cvt_bf16_T<<<cdiv(kH * 4 * kH, 256), 256, 0, stream>>>(cU1, cU1T, kH, 4 * kH);
  cvt_bf16_T<<<cdiv(kDIN * 4 * kH, 256), 256, 0, stream>>>(fW0, fW0T, kDIN, 4 * kH);
  cvt_bf16_T<<<cdiv(kH * 4 * kH, 256), 256, 0, stream>>>(fU0, fU0T, kH, 4 * kH);
  cvt_bf16_T<<<cdiv(kH * 4 * kH, 256), 256, 0, stream>>>(fW1, fW1T, kH, 4 * kH);
  cvt_bf16_T<<<cdiv(kH * 4 * kH, 256), 256, 0, stream>>>(fU1, fU1T, kH, 4 * kH);
  cvt_bf16_T<<<cdiv(kH * kPGV, 256), 256, 0, stream>>>(Wpage, WpageT, kH, kPGV);
  cvt_bf16_T<<<cdiv(kH * kOFFV, 256), 256, 0, stream>>>(Woff, WoffT, kH, kOFFV);

  // WvoT = WoT @ Wv^T  (C[n][k] = sum_j Wo[j][n] Wv[k][j] = (Wv@Wo)^T[n][k])
  gemm_bf16<<<dim3(1, kEPG / 32), 256, 0, stream>>>(WoT, kEPG, WvB, kEPG,
                                                    WvoT_f, kEPG, nullptr, kEPG, kEPG);
  cvt_bf16<<<cdiv(kEPG * kEPG, 256), 256, 0, stream>>>(WvoT_f, WvoT, kEPG * kEPG);
  compute_bvo<<<1, 256, 0, stream>>>(bv, Wo, bo, bvo);

  // ---- features (embeddings + fused attention) ------------------------------
  build_features<<<kTOK / 16, 256, 0, stream>>>(inputs, pc_emb, page_emb, off_emb,
                                                WqT, bq, WkB, WvoT, bvo, xfeat);

  // ---- two LSTM stacks (c: page head, f: offset head), run sequentially ----
  const bf16*  W0T_[2] = {cW0T, fW0T};
  const bf16*  U0T_[2] = {cU0T, fU0T};
  const bf16*  W1T_[2] = {cW1T, fW1T};
  const bf16*  U1T_[2] = {cU1T, fU1T};
  const float* b0_[2]  = {cb0, fb0};
  const float* b1_[2]  = {cb1, fb1};

  for (int sk = 0; sk < 2; ++sk) {
    // layer 0 input projections for all timesteps: (32768 x 576) @ (576 x 1024)
    gemm_bf16<<<dim3(cdiv(4 * kH, 512), kTOK / 32), 256, 0, stream>>>(
        xfeat, kDIN, W0T_[sk], kDIN, zbuf, 4 * kH, b0_[sk], 4 * kH, kDIN);
    for (int t = 0; t < kS; ++t)
      lstm_step<<<kB / 16, 256, 0, stream>>>(hs0, U0T_[sk], zbuf, cst, hs0, t);

    // layer 1 input projections: (32768 x 256) @ (256 x 1024)
    gemm_bf16<<<dim3(cdiv(4 * kH, 512), kTOK / 32), 256, 0, stream>>>(
        hs0, kH, W1T_[sk], kH, zbuf, 4 * kH, b1_[sk], 4 * kH, kH);
    for (int t = 0; t < kS; ++t)
      lstm_step<<<kB / 16, 256, 0, stream>>>(hs1, U1T_[sk], zbuf, cst, hs1, t);

    // head on h at t = S-1 (A rows strided by S*H through hs1)
    if (sk == 0) {
      gemm_bf16<<<dim3(cdiv(kPGV, 512), kB / 32), 256, 0, stream>>>(
          hs1 + (size_t)(kS - 1) * kH, kS * kH, WpageT, kH,
          out, kPGV + kOFFV, bpage, kPGV, kH);
    } else {
      gemm_bf16<<<dim3(1, kB / 32), 256, 0, stream>>>(
          hs1 + (size_t)(kS - 1) * kH, kS * kH, WoffT, kH,
          out + kPGV, kPGV + kOFFV, boff, kOFFV, kH);
    }
  }
}
